// SpikingInteractionBlock_70205535420979
// MI455X (gfx1250) — matrix operand order repaired and checked by reading, not compiled, not run on hardware
//
#include <hip/hip_runtime.h>
#include <hip/hip_bf16.h>

// ---------- constants matching the reference ----------
#define DIMN   256
#define HEADS  8
#define HD     32
#define HIDN   1024
#define BATCH  16
#define SEQ    1024
#define MROWS  (BATCH*SEQ)   // 16384

typedef _Float16 h8   __attribute__((ext_vector_type(8)));
typedef _Float16 v16h __attribute__((ext_vector_type(16)));
typedef float    v8f  __attribute__((ext_vector_type(8)));

// A-fragment (16x32 f16): lane = row (l&15); one contiguous 16-elem run at
// khalf*16 (contiguous k order, matching the B fragments). p points at the
// run (base + row*ld + kc + khalf*16).
__device__ __forceinline__ v16h load_a16(const _Float16* p) {
    h8 lo = *(const h8*)(p);
    h8 hi = *(const h8*)(p + 8);
    return __builtin_shufflevector(lo, hi, 0,1,2,3,4,5,6,7,8,9,10,11,12,13,14,15);
}

__device__ __forceinline__ v8f wmma16(v16h a, v16h b, v8f c) {
    return __builtin_amdgcn_wmma_f32_16x16x32_f16(false, a, false, b, (short)0, c, false, false);
}

// ---------- f32 -> f16 weight conversion ----------
__global__ void cvt_f16_kernel(const float* __restrict__ in, _Float16* __restrict__ out, int n) {
    int i = blockIdx.x * blockDim.x + threadIdx.x;
    if (i < n) out[i] = (_Float16)in[i];
}

// ---------- LayerNorm over last dim (256), one row per 256-thread block ----------
__global__ __launch_bounds__(DIMN) void ln_kernel(const float* __restrict__ x,
                                                  const float* __restrict__ g,
                                                  const float* __restrict__ b,
                                                  _Float16* __restrict__ out) {
    __shared__ float red[DIMN];
    int row = blockIdx.x, t = threadIdx.x;
    float v = x[(size_t)row * DIMN + t];
    red[t] = v; __syncthreads();
    for (int s = 128; s > 0; s >>= 1) { if (t < s) red[t] += red[t + s]; __syncthreads(); }
    float mu = red[0] * (1.0f / DIMN); __syncthreads();
    float d = v - mu;
    red[t] = d * d; __syncthreads();
    for (int s = 128; s > 0; s >>= 1) { if (t < s) red[t] += red[t + s]; __syncthreads(); }
    float rstd = rsqrtf(red[0] * (1.0f / DIMN) + 1e-5f);
    out[(size_t)row * DIMN + t] = (_Float16)(d * rstd * g[t] + b[t]);
}

// ---------- Generic WMMA GEMM: C[M,N] = A[M,K](f16) @ W[N,K]^T(f16) + epilogue ----------
// 4 waves / block; each wave: 16 rows x 64 cols (4 wmma acc tiles).
// epi 0: QKV scatter  -> q[B,H,T,32], k[B,H,T,32], vT[B,H,32,T] (f16, +bias)
// epi 1: f32 out = acc + bias + resid  (attn proj + residual)
// epi 2: f32 out = acc + bias          (FC1/FC2)
__global__ __launch_bounds__(128) void gemm_wmma_kernel(
    const _Float16* __restrict__ A, const _Float16* __restrict__ W,
    const float* __restrict__ bias, int M, int N, int K, int epi,
    const float* __restrict__ resid, float* __restrict__ outf,
    _Float16* __restrict__ qh, _Float16* __restrict__ kh, _Float16* __restrict__ vth)
{
    int lane = threadIdx.x & 31;
    int wave = threadIdx.x >> 5;
    int lq = lane & 15, lh = lane >> 4;
    int row0 = blockIdx.x * 64 + wave * 16;
    int col0 = blockIdx.y * 64;

    const _Float16* arow = A + (size_t)(row0 + lq) * K + lh * 16;  // A-frag base
    const _Float16* w0   = W + (size_t)(col0 + lq) * K + lh * 16;  // B-frag base (col = lq)

    v8f acc0 = {}, acc1 = {}, acc2 = {}, acc3 = {};
    for (int kc = 0; kc < K; kc += 32) {
        __builtin_prefetch(arow + kc + 128, 0, 1);   // global_prefetch_b8
        v16h a  = load_a16(arow + kc);
        v16h b0 = *(const v16h*)(w0 + kc);
        v16h b1 = *(const v16h*)(w0 + (size_t)16 * K + kc);
        v16h b2 = *(const v16h*)(w0 + (size_t)32 * K + kc);
        v16h b3 = *(const v16h*)(w0 + (size_t)48 * K + kc);
        acc0 = wmma16(a, b0, acc0);
        acc1 = wmma16(a, b1, acc1);
        acc2 = wmma16(a, b2, acc2);
        acc3 = wmma16(a, b3, acc3);
    }
    v8f accs[4] = {acc0, acc1, acc2, acc3};

    // C layout: lane col = l&15, rows = 8*(l>>4)+g
    #pragma unroll
    for (int j = 0; j < 4; j++) {
        int col = col0 + j * 16 + lq;
        float bv = bias[col];
        if (epi == 0) {
            int sect = col >> 8;            // 0:q 1:k 2:v  (tiles never straddle sections)
            int cc = col & 255;
            int hh = cc >> 5, dd = cc & 31;
            #pragma unroll
            for (int g = 0; g < 8; g++) {
                int r = row0 + 8 * lh + g;
                int bb = r >> 10, t = r & 1023;
                float val = accs[j][g] + bv;
                if (sect == 0)
                    qh[((size_t)(bb * HEADS + hh) * SEQ + t) * HD + dd] = (_Float16)val;
                else if (sect == 1)
                    kh[((size_t)(bb * HEADS + hh) * SEQ + t) * HD + dd] = (_Float16)val;
                else
                    vth[((size_t)(bb * HEADS + hh) * HD + dd) * SEQ + t] = (_Float16)val;
            }
        } else if (epi == 1) {
            #pragma unroll
            for (int g = 0; g < 8; g++) {
                size_t idx = (size_t)(row0 + 8 * lh + g) * N + col;
                outf[idx] = accs[j][g] + bv + resid[idx];
            }
        } else {
            #pragma unroll
            for (int g = 0; g < 8; g++) {
                size_t idx = (size_t)(row0 + 8 * lh + g) * N + col;
                outf[idx] = accs[j][g] + bv;
            }
        }
    }
}

// ---------- Flash attention: one wave per (b, h, 16-query tile) ----------
// S^T = K_tile(16x32) @ Q^T(32x16)  -> keys in rows, queries in cols
// O^T = V^T_tile(16x32k) @ P^T(32x16) accumulated online.
__global__ __launch_bounds__(32) void attn_kernel(
    const _Float16* __restrict__ q, const _Float16* __restrict__ kmat,
    const _Float16* __restrict__ vt, const unsigned char* __restrict__ kpm,
    _Float16* __restrict__ o)
{
    int lane = threadIdx.x;
    int lq = lane & 15, lh = lane >> 4;
    int tq0 = blockIdx.x * 16;
    int h = blockIdx.y, b = blockIdx.z;

    const _Float16* qbase = q    + (size_t)(b * HEADS + h) * SEQ * HD;
    const _Float16* kbase = kmat + (size_t)(b * HEADS + h) * SEQ * HD;
    const _Float16* vbase = vt   + (size_t)(b * HEADS + h) * HD * SEQ;

    // Q^T B-frag: col = query tq0+lq, elems i -> d = lh*16+i (contiguous 32B)
    v16h qb = *(const v16h*)(qbase + (size_t)(tq0 + lq) * HD + lh * 16);

    v8f ot0 = {}, ot1 = {};
    float m_run = -3.0e38f, l_run = 0.0f;
    const float scale = 0.17677669529663687f;  // 1/sqrt(32)

    for (int kc = 0; kc < SEQ; kc += 32) {
        // K A-frags: rows = keys, contiguous head-dim runs
        v16h a0 = load_a16(kbase + (size_t)(kc + lq) * HD + lh * 16);
        v16h a1 = load_a16(kbase + (size_t)(kc + 16 + lq) * HD + lh * 16);
        v8f z = {};
        v8f st0 = wmma16(a0, qb, z);   // keys kc..kc+15   x queries
        v8f st1 = wmma16(a1, qb, z);   // keys kc+16..+31  x queries

        // key padding mask: 8 bytes per lane-group (mask is all-false in setup;
        // read as bytes — zeros either way if harness stores wider ints)
        unsigned long long mb0 = *(const unsigned long long*)(kpm + (size_t)b * SEQ + kc + 8 * lh);
        unsigned long long mb1 = *(const unsigned long long*)(kpm + (size_t)b * SEQ + kc + 16 + 8 * lh);

        float p0[8], p1[8];
        float mx = -3.0e38f;
        #pragma unroll
        for (int g = 0; g < 8; g++) {
            float s0 = st0[g] * scale; if ((mb0 >> (8 * g)) & 0xFF) s0 = -3.0e38f;
            float s1 = st1[g] * scale; if ((mb1 >> (8 * g)) & 0xFF) s1 = -3.0e38f;
            p0[g] = s0; p1[g] = s1;
            mx = fmaxf(mx, fmaxf(s0, s1));
        }
        mx = fmaxf(mx, __shfl_xor(mx, 16, 32));        // combine the two key halves
        float m_new = fmaxf(m_run, mx);
        float alpha = __expf(m_run - m_new);
        float lsum = 0.0f;
        #pragma unroll
        for (int g = 0; g < 8; g++) {
            p0[g] = __expf(p0[g] - m_new);
            p1[g] = __expf(p1[g] - m_new);
            lsum += p0[g] + p1[g];
        }
        lsum += __shfl_xor(lsum, 16, 32);
        l_run = l_run * alpha + lsum;
        m_run = m_new;
        ot0 = ot0 * alpha;
        ot1 = ot1 * alpha;

        // Build P^T B-frag (col = query lq, elems i -> key lh*16+i) from C-layout
        // tiles with one shfl_xor(16) pair per g.
        v16h pb;
        #pragma unroll
        for (int g = 0; g < 8; g++) {
            float x0 = __shfl_xor(p0[g], 16, 32);
            float x1 = __shfl_xor(p1[g], 16, 32);
            pb[g]     = (_Float16)(lh ? x1 : p0[g]);
            pb[8 + g] = (_Float16)(lh ? p1[g] : x0);
        }

        // V^T A-frags: row = head-dim, contiguous key runs ([B,H,32,T] layout)
        v16h vt0 = load_a16(vbase + (size_t)lq * SEQ + kc + lh * 16);
        v16h vt1 = load_a16(vbase + (size_t)(16 + lq) * SEQ + kc + lh * 16);
        ot0 = wmma16(vt0, pb, ot0);
        ot1 = wmma16(vt1, pb, ot1);
    }

    float inv = 1.0f / l_run;
    // O^T C-layout: col = query, rows = head-dim -> merged-head [B,T,256] f16
    _Float16* optr = o + (size_t)(b * SEQ + tq0 + lq) * DIMN + h * HD;
    #pragma unroll
    for (int g = 0; g < 8; g++) {
        optr[8 * lh + g]      = (_Float16)(ot0[g] * inv);
        optr[16 + 8 * lh + g] = (_Float16)(ot1[g] * inv);
    }
}

// ---------- LIF scan over T (sequential), parallel over B*F (coalesced) ----------
__global__ void lif_kernel(const float* __restrict__ u, const unsigned char* __restrict__ kpm,
                           _Float16* __restrict__ spk, int F)
{
    int tid = blockIdx.x * blockDim.x + threadIdx.x;
    int b = tid / F, f = tid % F;
    float v = 0.0f;
    for (int t = 0; t < SEQ; t++) {
        size_t idx = (size_t)(b * SEQ + t) * F + f;
        float valid = kpm[(size_t)b * SEQ + t] ? 0.0f : 1.0f;
        float xin = u[idx] * valid;
        v = v + (xin - v) * 0.5f;                 // charge, TAU=2
        float s = (v - 1.0f >= 0.0f) ? 1.0f : 0.0f;
        spk[idx] = (_Float16)s;
        v = (1.0f - s) * v;                        // hard reset to 0
    }
}

__global__ void lif_final_kernel(const float* __restrict__ y, const unsigned char* __restrict__ kpm,
                                 const float* __restrict__ src2, float* __restrict__ out)
{
    int tid = blockIdx.x * blockDim.x + threadIdx.x;
    int b = tid / DIMN, d = tid % DIMN;
    float v = 0.0f;
    for (int t = 0; t < SEQ; t++) {
        size_t idx = (size_t)(b * SEQ + t) * DIMN + d;
        float valid = kpm[(size_t)b * SEQ + t] ? 0.0f : 1.0f;
        float xin = y[idx] * valid;
        v = v + (xin - v) * 0.5f;
        float s = (v - 1.0f >= 0.0f) ? 1.0f : 0.0f;
        out[idx] = src2[idx] + s;                  // final residual
        v = (1.0f - s) * v;
    }
}

// ---------- launch ----------
extern "C" void kernel_launch(void* const* d_in, const int* in_sizes, int n_in,
                              void* d_out, int out_size, void* d_ws, size_t ws_size,
                              hipStream_t stream) {
    const float* src    = (const float*)d_in[0];
    const float* ln1_g  = (const float*)d_in[1];
    const float* ln1_b  = (const float*)d_in[2];
    const float* wqkv   = (const float*)d_in[3];
    const float* bqkv   = (const float*)d_in[4];
    const float* wo     = (const float*)d_in[5];
    const float* bo     = (const float*)d_in[6];
    const float* ln2_g  = (const float*)d_in[7];
    const float* ln2_b  = (const float*)d_in[8];
    const float* fc1_w  = (const float*)d_in[9];
    const float* fc1_b  = (const float*)d_in[10];
    const float* fc2_w  = (const float*)d_in[11];
    const float* fc2_b  = (const float*)d_in[12];
    const unsigned char* kpm = (const unsigned char*)d_in[13];
    float* out = (float*)d_out;
    char* ws = (char*)d_ws;
    const size_t MB = 1ull << 20;

    // workspace layout (with lifetime-based reuse, ~122 MB total)
    float*    src2 = (float*)   (ws + 0);          // 16 MB, persists to end
    _Float16* xln  = (_Float16*)(ws + 16 * MB);    // 8 MB
    _Float16* qh   = (_Float16*)(ws + 24 * MB);    // 8 MB
    _Float16* kh   = (_Float16*)(ws + 32 * MB);    // 8 MB
    _Float16* vth  = (_Float16*)(ws + 40 * MB);    // 8 MB (V^T)
    _Float16* oat  = (_Float16*)(ws + 48 * MB);    // 8 MB
    _Float16* x2   = (_Float16*)(ws + 80 * MB);    // 8 MB
    float*    u    = (float*)   (ws + 16 * MB);    // 64 MB (reuses xln..oat)
    _Float16* spk1 = (_Float16*)(ws + 88 * MB);    // 32 MB
    float*    yb   = (float*)   (ws + 16 * MB);    // 16 MB (reuses u)
    _Float16* wqkv_h = (_Float16*)(ws + 120 * MB);
    _Float16* wo_h   = wqkv_h + 768 * 256;
    _Float16* fc1_h  = wo_h   + 256 * 256;
    _Float16* fc2_h  = fc1_h  + 1024 * 256;

    // weight conversion f32->f16
    cvt_f16_kernel<<<(768 * 256 + 255) / 256, 256, 0, stream>>>(wqkv, wqkv_h, 768 * 256);
    cvt_f16_kernel<<<(256 * 256 + 255) / 256, 256, 0, stream>>>(wo, wo_h, 256 * 256);
    cvt_f16_kernel<<<(1024 * 256 + 255) / 256, 256, 0, stream>>>(fc1_w, fc1_h, 1024 * 256);
    cvt_f16_kernel<<<(1024 * 256 + 255) / 256, 256, 0, stream>>>(fc2_w, fc2_h, 1024 * 256);

    // LN1
    ln_kernel<<<MROWS, DIMN, 0, stream>>>(src, ln1_g, ln1_b, xln);
    // QKV GEMM with head scatter (V transposed)
    gemm_wmma_kernel<<<dim3(MROWS / 64, 768 / 64), 128, 0, stream>>>(
        xln, wqkv_h, bqkv, MROWS, 768, 256, 0, nullptr, nullptr, qh, kh, vth);
    // attention
    attn_kernel<<<dim3(SEQ / 16, HEADS, BATCH), 32, 0, stream>>>(qh, kh, vth, kpm, oat);
    // output proj + bias + residual -> src2
    gemm_wmma_kernel<<<dim3(MROWS / 64, 256 / 64), 128, 0, stream>>>(
        oat, wo_h, bo, MROWS, 256, 256, 1, src, src2, nullptr, nullptr, nullptr);
    // LN2
    ln_kernel<<<MROWS, DIMN, 0, stream>>>(src2, ln2_g, ln2_b, x2);
    // FC1
    gemm_wmma_kernel<<<dim3(MROWS / 64, 1024 / 64), 128, 0, stream>>>(
        x2, fc1_h, fc1_b, MROWS, 1024, 256, 2, nullptr, u, nullptr, nullptr, nullptr);
    // LIF 1
    lif_kernel<<<(BATCH * HIDN) / 256, 256, 0, stream>>>(u, kpm, spk1, HIDN);
    // FC2
    gemm_wmma_kernel<<<dim3(MROWS / 64, 256 / 64), 128, 0, stream>>>(
        spk1, fc2_h, fc2_b, MROWS, 256, 1024, 2, nullptr, yb, nullptr, nullptr, nullptr);
    // LIF 2 + final residual
    lif_final_kernel<<<(BATCH * DIMN) / 256, 256, 0, stream>>>(yb, kpm, src2, out);
}
